// SparseMoELayer_89343909691603
// MI455X (gfx1250) — matrix operand order repaired and checked by reading, not compiled, not run on hardware
//
#include <hip/hip_runtime.h>

// ---------------------------------------------------------------------------
// Sparse MoE (top-2 of 8 experts), T=8192 tokens, D=1024, H=2048.
// Routed compute: 137 GFLOP bf16 WMMA; ~200MB HBM traffic -> mildly
// compute-bound on MI455X; v_wmma_f32_16x16x32_bf16 with async-LDS pipeline.
// LDS tiles padded to 80B row pitch (ROWP=40 bf16) for conflict-free b128 IO.
// ---------------------------------------------------------------------------

#define T_TOK 8192
#define DIM   1024
#define NEXP  8
#define HID   2048
#define NSLOT (2 * T_TOK)        // exactly K*T routed slots
#define SLOT_CAP (NSLOT + 128)   // pad for tile overrun
#define ROWP  40                 // LDS row pitch in bf16 elems (80B, 16B-aligned)

typedef unsigned short u16;
typedef __attribute__((ext_vector_type(8)))  __bf16 v8bf;
typedef __attribute__((ext_vector_type(16))) __bf16 v16bf;
typedef __attribute__((ext_vector_type(8)))  float  v8f;
typedef __attribute__((ext_vector_type(4)))  int    i4v;

#define AS1 __attribute__((address_space(1)))
#define AS3 __attribute__((address_space(3)))

#if __has_builtin(__builtin_amdgcn_global_load_async_to_lds_b128)
#define HAS_ASYNC_LDS 1
__device__ __forceinline__ void async_b128(const void* g, void* l) {
  // param0: int4 AS1* (global src), param1: int4 AS3* (LDS dst), imm offset, imm cpol
  __builtin_amdgcn_global_load_async_to_lds_b128(
      (AS1 i4v*)(void*)g, (AS3 i4v*)l, 0, 0);
}
__device__ __forceinline__ void wait_async0() {
#if __has_builtin(__builtin_amdgcn_s_wait_asynccnt)
  __builtin_amdgcn_s_wait_asynccnt(0);
#else
  asm volatile("s_wait_asynccnt 0x0" ::: "memory");
#endif
}
#else
#define HAS_ASYNC_LDS 0
#endif

__device__ __forceinline__ u16 f2bf(float f) {
  unsigned int u = __builtin_bit_cast(unsigned int, f);
  u += 0x7FFFu + ((u >> 16) & 1u);            // round-to-nearest-even
  return (u16)(u >> 16);
}

__device__ __forceinline__ v16bf mk16(v8bf lo, v8bf hi) {
  return __builtin_shufflevector(lo, hi, 0,1,2,3,4,5,6,7,8,9,10,11,12,13,14,15);
}

__device__ __forceinline__ v8f wmma_bf16(v16bf a, v16bf b, v8f c) {
  // (neg_a, A, neg_b, B, c_mod, C, reuse_a, reuse_b)
  return __builtin_amdgcn_wmma_f32_16x16x32_bf16(false, a, false, b, (short)0, c,
                                                 false, false);
}

// Fragment gather from LDS + 4x2 16x16x32 MMAs for one 32-wide K step.
// A frag (16x32 bf16): lane half kh -> K {kh*8..+7} U {16+kh*8..+7}.
// B frag (32x16 bf16): lane half kh -> K {kh*8..+7} U {16+kh*8..+7}.
__device__ __forceinline__ void frag_mma(const u16* __restrict__ As,
                                         const u16* __restrict__ Bs,
                                         int wm, int wn, int kh, int ln,
                                         v8f (&acc)[4][2]) {
  v16bf af[4], bfv[2];
  #pragma unroll
  for (int sm = 0; sm < 4; sm++) {
    const int m = wm * 64 + sm * 16 + ln;
    af[sm] = mk16(*(const v8bf*)&As[m * ROWP + kh * 8],
                  *(const v8bf*)&As[m * ROWP + 16 + kh * 8]);
  }
  #pragma unroll
  for (int sn = 0; sn < 2; sn++) {
    const int n = wn * 32 + sn * 16 + ln;
    bfv[sn] = mk16(*(const v8bf*)&Bs[n * ROWP + kh * 8],
                   *(const v8bf*)&Bs[n * ROWP + 16 + kh * 8]);
  }
  #pragma unroll
  for (int sm = 0; sm < 4; sm++)
    #pragma unroll
    for (int sn = 0; sn < 2; sn++)
      acc[sm][sn] = wmma_bf16(af[sm], bfv[sn], acc[sm][sn]);
}

// ---------------------------------------------------------------- utilities
__global__ __launch_bounds__(256) void zero_f_kernel(float* __restrict__ p, size_t n4) {
  size_t i = (size_t)blockIdx.x * blockDim.x + threadIdx.x;
  float4 z = make_float4(0.f, 0.f, 0.f, 0.f);
  for (; i < n4; i += (size_t)gridDim.x * blockDim.x)
    ((float4*)p)[i] = z;
}

__global__ void init_counters_kernel(int* __restrict__ counts, int* __restrict__ cursors) {
  int i = threadIdx.x;
  if (i < NEXP) { counts[i] = 0; cursors[i] = 0; }
}

__global__ __launch_bounds__(256) void cvt_x_kernel(const float* __restrict__ x,
                                                    u16* __restrict__ xb) {
  size_t i = ((size_t)blockIdx.x * blockDim.x + threadIdx.x) * 8;
  if (i >= (size_t)T_TOK * DIM) return;
  float4 a = *(const float4*)(x + i);
  float4 b = *(const float4*)(x + i + 4);
  u16 o[8] = { f2bf(a.x), f2bf(a.y), f2bf(a.z), f2bf(a.w),
               f2bf(b.x), f2bf(b.y), f2bf(b.z), f2bf(b.w) };
  *(uint4*)(xb + i) = *(const uint4*)o;
}

// src [E][R][C] fp32 -> dst [E][C][R] bf16  (makes GEMM K-dim contiguous)
__global__ __launch_bounds__(256) void transpose_cvt_kernel(const float* __restrict__ src,
                                                            u16* __restrict__ dst,
                                                            int R, int C) {
  __shared__ float tile[32][33];
  const int e  = blockIdx.z;
  const int r0 = blockIdx.x * 32;
  const int c0 = blockIdx.y * 32;
  const float* s = src + ((size_t)e * R + r0) * C + c0;
  for (int i = threadIdx.y; i < 32; i += 8)
    tile[i][threadIdx.x] = s[(size_t)i * C + threadIdx.x];
  __syncthreads();
  u16* d = dst + ((size_t)e * C + c0) * R + r0;
  for (int i = threadIdx.y; i < 32; i += 8)
    d[(size_t)i * R + threadIdx.x] = f2bf(tile[threadIdx.x][i]);
}

// ---------------------------------------------------------------- gating
__global__ __launch_bounds__(256) void gate_kernel(const float* __restrict__ x,
                                                   const float* __restrict__ gw,
                                                   const float* __restrict__ gb,
                                                   int* __restrict__ counts,
                                                   int2* __restrict__ topi,
                                                   float2* __restrict__ topw) {
  const int lane = threadIdx.x & 31;
  const int wv   = threadIdx.x >> 5;
  const int t    = blockIdx.x * 8 + wv;          // one wave32 per token
  const float* xr = x + (size_t)t * DIM;
  float s0=0,s1=0,s2=0,s3=0,s4=0,s5=0,s6=0,s7=0;
  for (int d = lane; d < DIM; d += 32) {
    const float xv = xr[d];
    const float4* g = (const float4*)(gw + d * NEXP);  // row of 8 floats, coalesced
    const float4 ga = g[0], gc = g[1];
    s0 += xv * ga.x; s1 += xv * ga.y; s2 += xv * ga.z; s3 += xv * ga.w;
    s4 += xv * gc.x; s5 += xv * gc.y; s6 += xv * gc.z; s7 += xv * gc.w;
  }
  float s[8] = { s0,s1,s2,s3,s4,s5,s6,s7 };
  #pragma unroll
  for (int e = 0; e < 8; e++)
    #pragma unroll
    for (int m = 16; m >= 1; m >>= 1)
      s[e] += __shfl_xor(s[e], m, 32);
  if (lane == 0) {
    float l[8], p[8];
    float mx = -1e30f;
    #pragma unroll
    for (int e = 0; e < 8; e++) { l[e] = s[e] + gb[e]; mx = fmaxf(mx, l[e]); }
    float sum = 0.f;
    #pragma unroll
    for (int e = 0; e < 8; e++) { p[e] = __expf(l[e] - mx); sum += p[e]; }
    int i0 = 0; float p0 = p[0];
    #pragma unroll
    for (int e = 1; e < 8; e++) if (p[e] > p0) { p0 = p[e]; i0 = e; }
    int i1 = (i0 == 0) ? 1 : 0; float p1 = p[i1];
    #pragma unroll
    for (int e = 0; e < 8; e++) if (e != i0 && p[e] > p1) { p1 = p[e]; i1 = e; }
    const float denom = p0 + p1 + 1e-9f * sum;   // matches reference renorm
    atomicAdd(&counts[i0], 1);
    atomicAdd(&counts[i1], 1);
    topi[t] = make_int2(i0, i1);
    topw[t] = make_float2(p0 / denom, p1 / denom);
  }
}

__global__ void prefix_kernel(const int* __restrict__ counts, int* __restrict__ offs,
                              int* __restrict__ cursors) {
  if (threadIdx.x == 0) {
    int a = 0;
    for (int e = 0; e < NEXP; e++) { offs[e] = a; cursors[e] = a; a += counts[e]; }
  }
}

__global__ __launch_bounds__(256) void scatter_kernel(const int2* __restrict__ topi,
                                                      const float2* __restrict__ topw,
                                                      int* __restrict__ cursors,
                                                      int* __restrict__ token_ids,
                                                      float* __restrict__ slot_w) {
  const int t = blockIdx.x * 256 + threadIdx.x;
  if (t >= T_TOK) return;
  const int2 i = topi[t];
  const float2 w = topw[t];
  int s0 = atomicAdd(&cursors[i.x], 1);
  token_ids[s0] = t; slot_w[s0] = w.x;
  int s1 = atomicAdd(&cursors[i.y], 1);
  token_ids[s1] = t; slot_w[s1] = w.y;
}

// ---------------------------------------------------------------- FFN GEMMs
// Tile: 128(M slots) x 128(N) x K, 8 waves; each wave owns 4x2 16x16 accs.
// Async path: double-buffered LDS fed by GLOBAL_LOAD_ASYNC_TO_LDS_B128,
// overlapped with WMMA; fallback: register-staged pipeline.

__global__ __launch_bounds__(256) void ffn1_kernel(const u16* __restrict__ xb,
                                                   const u16* __restrict__ w1t,
                                                   const float* __restrict__ b1,
                                                   const int* __restrict__ token_ids,
                                                   const int* __restrict__ counts,
                                                   const int* __restrict__ offs,
                                                   u16* __restrict__ hact) {
  const int e   = blockIdx.z;
  const int cnt = counts[e];
  const int m0  = blockIdx.x * 128;
  if (m0 >= cnt) return;                 // worst-case grid, early exit
  const int off = offs[e];
  const int n0  = blockIdx.y * 128;

  const int tid  = threadIdx.x;
  const int lane = tid & 31, wv = tid >> 5;
  const int wm = wv & 1, wn = wv >> 1;
  const int kh = lane >> 4, ln = lane & 15;

  const int lrow = tid >> 1, lseg = tid & 1;       // tile staging: 2 thr/row
  const int arow = m0 + lrow;
  const int tok  = (arow < cnt) ? token_ids[off + arow] : 0;
  const u16* aRow = xb  + (size_t)tok * DIM + lseg * 16;
  const u16* bRow = w1t + ((size_t)e * HID + n0 + lrow) * DIM + lseg * 16;
  const int sidx = lrow * ROWP + lseg * 16;

  const v8f vz = { 0.f,0.f,0.f,0.f,0.f,0.f,0.f,0.f };
  v8f acc[4][2];
  #pragma unroll
  for (int i = 0; i < 4; i++) { acc[i][0] = vz; acc[i][1] = vz; }

#if HAS_ASYNC_LDS
  __shared__ u16 As[2][128 * ROWP];
  __shared__ u16 Bs[2][128 * ROWP];
  async_b128(aRow,     &As[0][sidx]);
  async_b128(aRow + 8, &As[0][sidx + 8]);
  async_b128(bRow,     &Bs[0][sidx]);
  async_b128(bRow + 8, &Bs[0][sidx + 8]);
  wait_async0();
  __syncthreads();
  for (int kk = 0, it = 0; kk < DIM; kk += 32, ++it) {
    const int cur = it & 1;
    if (kk + 32 < DIM) {
      const int nxt = cur ^ 1;
      async_b128(aRow + kk + 32,     &As[nxt][sidx]);
      async_b128(aRow + kk + 40,     &As[nxt][sidx + 8]);
      async_b128(bRow + kk + 32,     &Bs[nxt][sidx]);
      async_b128(bRow + kk + 40,     &Bs[nxt][sidx + 8]);
    }
    frag_mma(As[cur], Bs[cur], wm, wn, kh, ln, acc);
    wait_async0();
    __syncthreads();
  }
#else
  __shared__ u16 As[128 * ROWP];
  __shared__ u16 Bs[128 * ROWP];
  uint4 ra0 = *(const uint4*)(aRow);
  uint4 ra1 = *(const uint4*)(aRow + 8);
  uint4 rb0 = *(const uint4*)(bRow);
  uint4 rb1 = *(const uint4*)(bRow + 8);
  for (int kk = 0; kk < DIM; kk += 32) {
    __syncthreads();
    *(uint4*)&As[sidx]     = ra0;
    *(uint4*)&As[sidx + 8] = ra1;
    *(uint4*)&Bs[sidx]     = rb0;
    *(uint4*)&Bs[sidx + 8] = rb1;
    __syncthreads();
    if (kk + 32 < DIM) {
      ra0 = *(const uint4*)(aRow + kk + 32);
      ra1 = *(const uint4*)(aRow + kk + 40);
      rb0 = *(const uint4*)(bRow + kk + 32);
      rb1 = *(const uint4*)(bRow + kk + 40);
      __builtin_prefetch(aRow + kk + 64, 0, 3);
      __builtin_prefetch(bRow + kk + 64, 0, 3);
    }
    frag_mma(As, Bs, wm, wn, kh, ln, acc);
  }
#endif

  const float* b1e = b1 + e * HID;
  #pragma unroll
  for (int sn = 0; sn < 2; sn++) {
    const int n = n0 + wn * 32 + sn * 16 + ln;
    const float bb = b1e[n];
    #pragma unroll
    for (int sm = 0; sm < 4; sm++) {
      const int mBase = wm * 64 + sm * 16 + kh * 8;   // C: m = kh*8 + vgpr
      #pragma unroll
      for (int r = 0; r < 8; r++) {
        const int m = m0 + mBase + r;
        if (m < cnt) {
          float v = acc[sm][sn][r] + bb;
          v = v > 0.f ? v : 0.f;                       // ReLU
          hact[(size_t)(off + m) * HID + n] = f2bf(v);
        }
      }
    }
  }
}

__global__ __launch_bounds__(256) void ffn2_kernel(const u16* __restrict__ hact,
                                                   const u16* __restrict__ w2t,
                                                   const float* __restrict__ b2,
                                                   const int* __restrict__ token_ids,
                                                   const float* __restrict__ slot_w,
                                                   const int* __restrict__ counts,
                                                   const int* __restrict__ offs,
                                                   float* __restrict__ out) {
  const int e   = blockIdx.z;
  const int cnt = counts[e];
  const int m0  = blockIdx.x * 128;
  if (m0 >= cnt) return;
  const int off = offs[e];
  const int n0  = blockIdx.y * 128;

  const int tid  = threadIdx.x;
  const int lane = tid & 31, wv = tid >> 5;
  const int wm = wv & 1, wn = wv >> 1;
  const int kh = lane >> 4, ln = lane & 15;

  const int lrow = tid >> 1, lseg = tid & 1;
  const u16* aRow = hact + (size_t)(off + m0 + lrow) * HID + lseg * 16;
  const u16* bRow = w2t + ((size_t)e * DIM + n0 + lrow) * HID + lseg * 16;
  const int sidx = lrow * ROWP + lseg * 16;

  const v8f vz = { 0.f,0.f,0.f,0.f,0.f,0.f,0.f,0.f };
  v8f acc[4][2];
  #pragma unroll
  for (int i = 0; i < 4; i++) { acc[i][0] = vz; acc[i][1] = vz; }

#if HAS_ASYNC_LDS
  __shared__ u16 As[2][128 * ROWP];
  __shared__ u16 Bs[2][128 * ROWP];
  async_b128(aRow,     &As[0][sidx]);
  async_b128(aRow + 8, &As[0][sidx + 8]);
  async_b128(bRow,     &Bs[0][sidx]);
  async_b128(bRow + 8, &Bs[0][sidx + 8]);
  wait_async0();
  __syncthreads();
  for (int kk = 0, it = 0; kk < HID; kk += 32, ++it) {
    const int cur = it & 1;
    if (kk + 32 < HID) {
      const int nxt = cur ^ 1;
      async_b128(aRow + kk + 32,     &As[nxt][sidx]);
      async_b128(aRow + kk + 40,     &As[nxt][sidx + 8]);
      async_b128(bRow + kk + 32,     &Bs[nxt][sidx]);
      async_b128(bRow + kk + 40,     &Bs[nxt][sidx + 8]);
    }
    frag_mma(As[cur], Bs[cur], wm, wn, kh, ln, acc);
    wait_async0();
    __syncthreads();
  }
#else
  __shared__ u16 As[128 * ROWP];
  __shared__ u16 Bs[128 * ROWP];
  uint4 ra0 = *(const uint4*)(aRow);
  uint4 ra1 = *(const uint4*)(aRow + 8);
  uint4 rb0 = *(const uint4*)(bRow);
  uint4 rb1 = *(const uint4*)(bRow + 8);
  for (int kk = 0; kk < HID; kk += 32) {
    __syncthreads();
    *(uint4*)&As[sidx]     = ra0;
    *(uint4*)&As[sidx + 8] = ra1;
    *(uint4*)&Bs[sidx]     = rb0;
    *(uint4*)&Bs[sidx + 8] = rb1;
    __syncthreads();
    if (kk + 32 < HID) {
      ra0 = *(const uint4*)(aRow + kk + 32);
      ra1 = *(const uint4*)(aRow + kk + 40);
      rb0 = *(const uint4*)(bRow + kk + 32);
      rb1 = *(const uint4*)(bRow + kk + 40);
      __builtin_prefetch(aRow + kk + 64, 0, 3);
      __builtin_prefetch(bRow + kk + 64, 0, 3);
    }
    frag_mma(As, Bs, wm, wn, kh, ln, acc);
  }
#endif

  const float* b2e = b2 + e * DIM;
  const int nA = n0 + wn * 32 + ln;
  const float bb0 = b2e[nA];
  const float bb1 = b2e[nA + 16];
  #pragma unroll
  for (int sm = 0; sm < 4; sm++) {
    const int mBase = wm * 64 + sm * 16 + kh * 8;
    #pragma unroll
    for (int r = 0; r < 8; r++) {
      const int m = m0 + mBase + r;
      if (m < cnt) {
        const int slot = off + m;
        const int tok  = token_ids[slot];
        const float w  = slot_w[slot];
        unsafeAtomicAdd(&out[(size_t)tok * DIM + nA],      (acc[sm][0][r] + bb0) * w);
        unsafeAtomicAdd(&out[(size_t)tok * DIM + nA + 16], (acc[sm][1][r] + bb1) * w);
      }
    }
  }
}

// ------------------------------------------------ TDM compile probe (unlaunched)
// Hand-packed D# per CDNA5 ISA: 2D tensor, 2-byte elements,
// 32x128 tile, dim0 stride = row pitch. This toolchain exposes the 6-arg
// tensor_load_to_lds (g0, g1, g2, g3, g4, cpol); extra group passed as zeros.
#if __has_builtin(__builtin_amdgcn_tensor_load_to_lds)
typedef __attribute__((ext_vector_type(4))) unsigned int v4u;
typedef __attribute__((ext_vector_type(8))) int v8i;
typedef __attribute__((ext_vector_type(4))) int v4i;

__global__ void tdm_probe_kernel(const u16* __restrict__ src, u16* __restrict__ dst,
                                 int cols, int rows) {
  __shared__ u16 buf[128 * 32];
  const unsigned long long ga = (unsigned long long)(const void*)src;
  const unsigned td0 = (unsigned)cols, td1 = (unsigned)rows;
  const unsigned long long st0 = (unsigned long long)(unsigned)cols;
  v4u g0 = { 1u,                                   // count=1 (valid descriptor)
             0u,                                   // lds_addr (probe: base)
             (unsigned)ga,                         // global addr lo
             (unsigned)((ga >> 32) & 0x01FFFFFFull) | (2u << 30) };  // hi|type=2
  v8i g1 = { (int)(1u << 16),                      // data_size=1 (2 bytes)
             (int)((td0 & 0xFFFFu) << 16),         // tensor_dim0[15:0]
             (int)((td0 >> 16) | ((td1 & 0xFFFFu) << 16)),  // dim0 hi | dim1 lo
             (int)((td1 >> 16) | (32u << 16)),     // dim1 hi | tile_dim0=32
             (int)128,                             // tile_dim1=128, tile_dim2=0
             (int)(unsigned)(st0 & 0xFFFFFFFFull), // dim0_stride lo32
             (int)(unsigned)(st0 >> 32),           // dim0_stride hi16 | dim1_stride lo16
             0 };
  v4i g2 = { 0, 0, 0, 0 };
  v4i g3 = { 0, 0, 0, 0 };
  v8i g4 = { 0, 0, 0, 0, 0, 0, 0, 0 };
  if ((threadIdx.x >> 5) == 0) {                   // one wave issues the DMA
    __builtin_amdgcn_tensor_load_to_lds(g0, g1, g2, g3, g4, 0);
#if __has_builtin(__builtin_amdgcn_s_wait_tensorcnt)
    __builtin_amdgcn_s_wait_tensorcnt(0);
#else
    asm volatile("s_wait_tensorcnt 0x0" ::: "memory");
#endif
  }
  __syncthreads();
  dst[threadIdx.x] = buf[threadIdx.x];
}
#endif

// ---------------------------------------------------------------- launcher
extern "C" void kernel_launch(void* const* d_in, const int* in_sizes, int n_in,
                              void* d_out, int out_size, void* d_ws, size_t ws_size,
                              hipStream_t stream) {
  const float* x  = (const float*)d_in[0];
  const float* gw = (const float*)d_in[1];
  const float* gb = (const float*)d_in[2];
  const float* w1 = (const float*)d_in[3];
  const float* b1 = (const float*)d_in[4];
  const float* w2 = (const float*)d_in[5];
  const float* b2 = (const float*)d_in[6];
  float* out = (float*)d_out;

  // workspace carve-out (~150 MB)
  char* ws = (char*)d_ws;
  size_t off = 0;
  auto carve = [&](size_t bytes) -> char* {
    char* p = ws + off;
    off = (off + bytes + 255) & ~(size_t)255;
    return p;
  };
  u16*    xb        = (u16*)   carve((size_t)T_TOK * DIM * 2);
  u16*    w1t       = (u16*)   carve((size_t)NEXP * HID * DIM * 2);
  u16*    w2t       = (u16*)   carve((size_t)NEXP * DIM * HID * 2);
  u16*    hact      = (u16*)   carve((size_t)SLOT_CAP * HID * 2);
  int*    token_ids = (int*)   carve((size_t)SLOT_CAP * 4);
  float*  slot_w    = (float*) carve((size_t)SLOT_CAP * 4);
  int2*   topi      = (int2*)  carve((size_t)T_TOK * 8);
  float2* topw      = (float2*)carve((size_t)T_TOK * 8);
  int*    counts    = (int*)   carve(NEXP * 4);
  int*    offsp     = (int*)   carve(NEXP * 4);
  int*    cursors   = (int*)   carve(NEXP * 4);
  (void)ws_size; (void)in_sizes; (void)n_in; (void)out_size;

  // init
  zero_f_kernel<<<2048, 256, 0, stream>>>(out, (size_t)T_TOK * DIM / 4);
  init_counters_kernel<<<1, 32, 0, stream>>>(counts, cursors);

  // precision conversion / weight transposes (K-contiguous layouts for WMMA)
  cvt_x_kernel<<<(T_TOK * DIM / 8 + 255) / 256, 256, 0, stream>>>(x, xb);
  dim3 tb(32, 8);
  transpose_cvt_kernel<<<dim3(DIM / 32, HID / 32, NEXP), tb, 0, stream>>>(w1, w1t, DIM, HID);
  transpose_cvt_kernel<<<dim3(HID / 32, DIM / 32, NEXP), tb, 0, stream>>>(w2, w2t, HID, DIM);

  // routing
  gate_kernel<<<T_TOK / 8, 256, 0, stream>>>(x, gw, gb, counts, topi, topw);
  prefix_kernel<<<1, 1, 0, stream>>>(counts, offsp, cursors);
  scatter_kernel<<<T_TOK / 256, 256, 0, stream>>>(topi, topw, cursors, token_ids, slot_w);

  // routed expert FFN (worst-case grids; blocks early-exit past per-expert count)
  ffn1_kernel<<<dim3(T_TOK / 128, HID / 128, NEXP), 256, 0, stream>>>(
      xb, w1t, b1, token_ids, counts, offsp, hact);
  ffn2_kernel<<<dim3(T_TOK / 128, DIM / 128, NEXP), 256, 0, stream>>>(
      hact, w2t, b2, token_ids, slot_w, counts, offsp, out);
}
